// MultiHeadedAttention_63282048139552
// MI455X (gfx1250) — matrix operand order repaired and checked by reading, not compiled, not run on hardware
//
#include <hip/hip_runtime.h>

// ---------------------------------------------------------------------------
// CDNA5 (gfx1250) multi-headed attention, bf16 WMMA with f32 accumulation.
//   Stage 0: one-shot fp32->bf16 conversion of Wq/Wk/Wv (kills redundant cvt)
//   Stage 1: fused QKV projection GEMMs  (v_wmma_f32_16x16x32_bf16),
//            32(M) x 64(N) tile per wave, bf16 B-fragments straight from ws
//   Stage 2: flash-attention (online softmax), WMMA for QK^T and PV
// ---------------------------------------------------------------------------

typedef __attribute__((ext_vector_type(16))) __bf16 v16bf;
typedef __attribute__((ext_vector_type(8)))  float  v8f;

#define LOG2E 1.4426950408889634f

// ----- WMMA fragment loaders ------------------------------------------------
// A matrix (16x32 bf16): lane l holds row m=l&15; element j maps to
// K = (l>>4)*8 + (j<8 ? j : 8+j)  -> two contiguous 16B chunks at +0 and +16.
static __device__ __forceinline__ v16bf
load_frag_a_bf16(const __bf16* base, int ld, int m0, int k0, int lane) {
  const int m  = lane & 15;
  const int kb = (lane >> 4) * 8;
  const __bf16* p = base + (size_t)(m0 + m) * ld + (k0 + kb);
  union { v16bf v; uint4 q[2]; } u;
  u.q[0] = *(const uint4*)(p);
  u.q[1] = *(const uint4*)(p + 16);
  return u.v;
}

// B matrix (32x16 bf16): lane l holds column n=l&15; element j maps to
// K = (l>>4)*8 + (j<8 ? j : 8+j)  -> two contiguous 16B chunks at +0 and +16.
static __device__ __forceinline__ v16bf
load_frag_b_bf16(const __bf16* base, int ld, int n0, int k0, int lane) {
  const int n  = lane & 15;
  const int kb = (lane >> 4) * 8;
  const __bf16* p = base + (size_t)(n0 + n) * ld + (k0 + kb);
  union { v16bf v; uint4 q[2]; } u;
  u.q[0] = *(const uint4*)(p);
  u.q[1] = *(const uint4*)(p + 16);
  return u.v;
}

static __device__ __forceinline__ v16bf
cvt16(float4 f0, float4 f1, float4 f2, float4 f3) {
  v16bf v;
  v[0]  = (__bf16)f0.x; v[1]  = (__bf16)f0.y; v[2]  = (__bf16)f0.z; v[3]  = (__bf16)f0.w;
  v[4]  = (__bf16)f1.x; v[5]  = (__bf16)f1.y; v[6]  = (__bf16)f1.z; v[7]  = (__bf16)f1.w;
  v[8]  = (__bf16)f2.x; v[9]  = (__bf16)f2.y; v[10] = (__bf16)f2.z; v[11] = (__bf16)f2.w;
  v[12] = (__bf16)f3.x; v[13] = (__bf16)f3.y; v[14] = (__bf16)f3.z; v[15] = (__bf16)f3.w;
  return v;
}

// A fragment from fp32 source (global X, or LDS P tile), converting to bf16.
static __device__ __forceinline__ v16bf
load_frag_a_f32(const float* base, int ld, int m0, int k0, int lane) {
  const int m  = lane & 15;
  const int kb = (lane >> 4) * 8;
  const float* p = base + (size_t)(m0 + m) * ld + (k0 + kb);
  float4 f0 = *(const float4*)(p);
  float4 f1 = *(const float4*)(p + 4);
  float4 f2 = *(const float4*)(p + 16);
  float4 f3 = *(const float4*)(p + 20);
  return cvt16(f0, f1, f2, f3);
}

static __device__ __forceinline__ v8f vzero8() {
  v8f z = {0.f, 0.f, 0.f, 0.f, 0.f, 0.f, 0.f, 0.f};
  return z;
}

// ---------------------------------------------------------------------------
// Stage 0: fp32 -> bf16 conversion of the three weight matrices.
// grid = (512, 3); block = 256; 8 elements per thread (b128 in, b128 out).
// ---------------------------------------------------------------------------
__global__ __launch_bounds__(256) void mha_cvt_w_kernel(
    const float* __restrict__ Wq, const float* __restrict__ Wk,
    const float* __restrict__ Wv,
    void* Wqo, void* Wko, void* Wvo) {
  const float* src;
  __bf16* dst;
  if (blockIdx.y == 0)      { src = Wq; dst = (__bf16*)Wqo; }
  else if (blockIdx.y == 1) { src = Wk; dst = (__bf16*)Wko; }
  else                      { src = Wv; dst = (__bf16*)Wvo; }

  const size_t i = ((size_t)blockIdx.x * 256 + threadIdx.x) * 8;
  float4 f0 = *(const float4*)(src + i);
  float4 f1 = *(const float4*)(src + i + 4);
  union { __bf16 h[8]; uint4 q; } u;
  u.h[0] = (__bf16)f0.x; u.h[1] = (__bf16)f0.y; u.h[2] = (__bf16)f0.z; u.h[3] = (__bf16)f0.w;
  u.h[4] = (__bf16)f1.x; u.h[5] = (__bf16)f1.y; u.h[6] = (__bf16)f1.z; u.h[7] = (__bf16)f1.w;
  *(uint4*)(dst + i) = u.q;
}

// ---------------------------------------------------------------------------
// Stage 1: Y = X @ W^T + b.  grid = (32, 16, 3); block = 256 (8 waves).
// Each wave -> 32(M) x 64(N) output tile, K loop of 32; W read as bf16.
// z==0: Q (scaled by 1/sqrt(d_k)=0.125) -> [b,h,s,64] bf16
// z==1: K                               -> [b,h,s,64] bf16
// z==2: V (transposed)                  -> [b,h,64,s] bf16
// ---------------------------------------------------------------------------
__global__ __launch_bounds__(256) void mha_proj_kernel(
    const float* __restrict__ query, const float* __restrict__ key_,
    const float* __restrict__ value,
    const void* __restrict__ Wqb, const void* __restrict__ Wkb,
    const void* __restrict__ Wvb,
    const float* __restrict__ bq, const float* __restrict__ bk,
    const float* __restrict__ bv,
    void* Qo, void* Ko, void* Vo) {
  const int lane = threadIdx.x & 31;
  const int wave = threadIdx.x >> 5;
  const int which = blockIdx.z;

  const float* X;  const __bf16* W;  const float* bias;  __bf16* out;
  float scale = 1.0f;
  if (which == 0)      { X = query; W = (const __bf16*)Wqb; bias = bq; out = (__bf16*)Qo; scale = 0.125f; }
  else if (which == 1) { X = key_;  W = (const __bf16*)Wkb; bias = bk; out = (__bf16*)Ko; }
  else                 { X = value; W = (const __bf16*)Wvb; bias = bv; out = (__bf16*)Vo; }

  const int row0 = blockIdx.x * 256 + wave * 32;   // over B*S = 8192
  const int col0 = blockIdx.y * 64;                // over D_OUT = 1024

  v8f acc[2][4] = { { vzero8(), vzero8(), vzero8(), vzero8() },
                    { vzero8(), vzero8(), vzero8(), vzero8() } };

  for (int k0 = 0; k0 < 1024; k0 += 32) {
    const v16bf a0 = load_frag_a_f32(X, 1024, row0,      k0, lane);
    const v16bf a1 = load_frag_a_f32(X, 1024, row0 + 16, k0, lane);
#pragma unroll
    for (int nt = 0; nt < 4; ++nt) {
      const v16bf bfr = load_frag_b_bf16(W, 1024, col0 + nt * 16, k0, lane);
      acc[0][nt] = __builtin_amdgcn_wmma_f32_16x16x32_bf16(
          false, a0, false, bfr, (short)0, acc[0][nt], false, false);
      acc[1][nt] = __builtin_amdgcn_wmma_f32_16x16x32_bf16(
          false, a1, false, bfr, (short)0, acc[1][nt], false, false);
    }
  }

  const int half = lane >> 4;
  const int n16  = lane & 15;
#pragma unroll
  for (int mt = 0; mt < 2; ++mt) {
#pragma unroll
    for (int nt = 0; nt < 4; ++nt) {
      const int o = col0 + nt * 16 + n16;   // output channel
      const float bval = bias[o];
      const int h = o >> 6, d = o & 63;
#pragma unroll
      for (int r = 0; r < 8; ++r) {
        const int row = row0 + mt * 16 + r + half * 8;   // flattened (b,s)
        const int b = row >> 10, s = row & 1023;
        const float y = (acc[mt][nt][r] + bval) * scale;
        size_t idx;
        if (which == 2) idx = ((((size_t)b * 16 + h) * 64 + d) * 1024) + s;   // V^T
        else            idx = ((((size_t)b * 16 + h) * 1024 + s) * 64) + d;   // Q/K
        out[idx] = (__bf16)y;
      }
    }
  }
}

// ---------------------------------------------------------------------------
// Stage 2: flash attention.  grid = (8, 16, 8); block = 256 (8 waves).
// Wave owns 16 query rows x d_k=64; iterates kv in tiles of 64 with
// online softmax.  P transposed C-layout -> A-layout via padded LDS.
// ---------------------------------------------------------------------------
__global__ __launch_bounds__(256) void mha_attn_kernel(
    const void* __restrict__ Qp, const void* __restrict__ Kp,
    const void* __restrict__ Vtp, float* __restrict__ out) {
  __shared__ float plds[8][16][68];   // 68-float pitch: conflict-free b128 reads

  const int lane = threadIdx.x & 31;
  const int wave = threadIdx.x >> 5;
  const int h = blockIdx.y, b = blockIdx.z;
  const int row0 = blockIdx.x * 128 + wave * 16;   // query row block
  const int half = lane >> 4;
  const int n16  = lane & 15;

  const __bf16* Q  = (const __bf16*)Qp  + ((size_t)b * 16 + h) * 1024 * 64;
  const __bf16* K  = (const __bf16*)Kp  + ((size_t)b * 16 + h) * 1024 * 64;
  const __bf16* Vt = (const __bf16*)Vtp + ((size_t)b * 16 + h) * 64 * 1024;

  // Q fragments for k=0..31 and k=32..63 (loaded once; scale pre-folded)
  const v16bf qf0 = load_frag_a_bf16(Q, 64, row0, 0, lane);
  const v16bf qf1 = load_frag_a_bf16(Q, 64, row0, 32, lane);

  v8f o_acc[4] = { vzero8(), vzero8(), vzero8(), vzero8() };
  float m_r[8], l_r[8];
#pragma unroll
  for (int r = 0; r < 8; ++r) { m_r[r] = -__builtin_inff(); l_r[r] = 0.f; }

  float* myP = &plds[wave][0][0];

  for (int kv0 = 0; kv0 < 1024; kv0 += 64) {
    if (kv0 + 64 < 1024) {   // gfx1250 global_prefetch of next K/V tiles
      __builtin_prefetch(K  + (size_t)(kv0 + 64 + n16) * 64, 0, 0);
      __builtin_prefetch(Vt + (size_t)n16 * 1024 + kv0 + 64, 0, 0);
    }

    // ---- scores S = (Q*scale) . K^T over this 64-wide kv tile -------------
    v8f s_[4];
#pragma unroll
    for (int nt = 0; nt < 4; ++nt) {
      v8f c = vzero8();
      v16bf kb0 = load_frag_b_bf16(K, 64, kv0 + nt * 16, 0, lane);
      c = __builtin_amdgcn_wmma_f32_16x16x32_bf16(false, qf0, false, kb0,
                                                  (short)0, c, false, false);
      v16bf kb1 = load_frag_b_bf16(K, 64, kv0 + nt * 16, 32, lane);
      c = __builtin_amdgcn_wmma_f32_16x16x32_bf16(false, qf1, false, kb1,
                                                  (short)0, c, false, false);
      s_[nt] = c;
    }

    // ---- online softmax: row max across 4 subtiles and 16 lanes -----------
    float tmax[8];
#pragma unroll
    for (int r = 0; r < 8; ++r)
      tmax[r] = fmaxf(fmaxf(s_[0][r], s_[1][r]), fmaxf(s_[2][r], s_[3][r]));
#pragma unroll
    for (int off = 1; off < 16; off <<= 1) {
#pragma unroll
      for (int r = 0; r < 8; ++r)
        tmax[r] = fmaxf(tmax[r], __shfl_xor(tmax[r], off, 32));
    }

    float alpha[8], psum[8];
#pragma unroll
    for (int r = 0; r < 8; ++r) {
      const float mn = fmaxf(m_r[r], tmax[r]);
      alpha[r] = __builtin_exp2f((m_r[r] - mn) * LOG2E);
      m_r[r] = mn;
      psum[r] = 0.f;
    }
#pragma unroll
    for (int nt = 0; nt < 4; ++nt)
#pragma unroll
      for (int r = 0; r < 8; ++r) {
        const float p = __builtin_exp2f((s_[nt][r] - m_r[r]) * LOG2E);
        s_[nt][r] = p;
        psum[r] += p;
      }
#pragma unroll
    for (int off = 1; off < 16; off <<= 1) {
#pragma unroll
      for (int r = 0; r < 8; ++r) psum[r] += __shfl_xor(psum[r], off, 32);
    }
#pragma unroll
    for (int r = 0; r < 8; ++r) l_r[r] = l_r[r] * alpha[r] + psum[r];
#pragma unroll
    for (int nt = 0; nt < 4; ++nt)
#pragma unroll
      for (int r = 0; r < 8; ++r) o_acc[nt][r] *= alpha[r];

    // ---- transpose P (C-layout) -> A-layout via per-wave LDS slice --------
#pragma unroll
    for (int nt = 0; nt < 4; ++nt)
#pragma unroll
      for (int r = 0; r < 8; ++r)
        plds[wave][r + half * 8][nt * 16 + n16] = s_[nt][r];
    asm volatile("s_wait_dscnt 0" ::: "memory");   // wave-private LDS RAW fence

    const v16bf pf0 = load_frag_a_f32(myP, 68, 0, 0, lane);
    const v16bf pf1 = load_frag_a_f32(myP, 68, 0, 32, lane);

    // ---- O += P . V  (V stored transposed: [64][1024]) --------------------
#pragma unroll
    for (int nt = 0; nt < 4; ++nt) {
      v16bf vb0 = load_frag_b_bf16(Vt, 1024, nt * 16, kv0, lane);
      o_acc[nt] = __builtin_amdgcn_wmma_f32_16x16x32_bf16(
          false, pf0, false, vb0, (short)0, o_acc[nt], false, false);
      v16bf vb1 = load_frag_b_bf16(Vt, 1024, nt * 16, kv0 + 32, lane);
      o_acc[nt] = __builtin_amdgcn_wmma_f32_16x16x32_bf16(
          false, pf1, false, vb1, (short)0, o_acc[nt], false, false);
    }
  }

  // ---- epilogue: normalize and store in [b, s, h, d] flattening -----------
  float rinv[8];
#pragma unroll
  for (int r = 0; r < 8; ++r) rinv[r] = 1.0f / l_r[r];
#pragma unroll
  for (int nt = 0; nt < 4; ++nt) {
    const int d = nt * 16 + n16;
#pragma unroll
    for (int r = 0; r < 8; ++r) {
      const int s = row0 + r + half * 8;
      out[(((size_t)b * 1024 + s) * 16 + h) * 64 + d] = o_acc[nt][r] * rinv[r];
    }
  }
}

// ---------------------------------------------------------------------------
extern "C" void kernel_launch(void* const* d_in, const int* in_sizes, int n_in,
                              void* d_out, int out_size, void* d_ws, size_t ws_size,
                              hipStream_t stream) {
  (void)in_sizes; (void)n_in; (void)out_size; (void)ws_size;
  const float* query = (const float*)d_in[0];
  const float* key_  = (const float*)d_in[1];
  const float* value = (const float*)d_in[2];
  const float* Wq = (const float*)d_in[3];
  const float* bq = (const float*)d_in[4];
  const float* Wk = (const float*)d_in[5];
  const float* bk = (const float*)d_in[6];
  const float* Wv = (const float*)d_in[7];
  const float* bv = (const float*)d_in[8];

  char* ws = (char*)d_ws;
  const size_t seg  = (size_t)8 * 16 * 1024 * 64 * 2;   // 16 MiB per bf16 tensor
  const size_t wseg = (size_t)1024 * 1024 * 2;          // 2 MiB per bf16 weight
  void* Qb  = (void*)(ws);
  void* Kb  = (void*)(ws + seg);
  void* Vb  = (void*)(ws + 2 * seg);
  void* Wqb = (void*)(ws + 3 * seg);
  void* Wkb = (void*)(ws + 3 * seg + wseg);
  void* Wvb = (void*)(ws + 3 * seg + 2 * wseg);

  dim3 cgrid(512, 3), cblock(256);
  mha_cvt_w_kernel<<<cgrid, cblock, 0, stream>>>(Wq, Wk, Wv, Wqb, Wkb, Wvb);

  dim3 pgrid(32, 16, 3), pblock(256);
  mha_proj_kernel<<<pgrid, pblock, 0, stream>>>(query, key_, value,
                                                Wqb, Wkb, Wvb, bq, bk, bv,
                                                Qb, Kb, Vb);

  dim3 agrid(8, 16, 8), ablock(256);
  mha_attn_kernel<<<agrid, ablock, 0, stream>>>(Qb, Kb, Vb, (float*)d_out);
}
